// DP_40097814676059
// MI455X (gfx1250) — compile-verified, matrix operand order checked by reading, not run on hardware
//
#include <hip/hip_runtime.h>
#include <hip/hip_bf16.h>
#include <math.h>

// ---------------------------------------------------------------------------
// Problem constants (match reference)
// ---------------------------------------------------------------------------
#define NTYPES 2
#define NATOMS 1024
#define NBATCH 4
#define MAXNB  100
#define KROWS  112            // neighbors padded to 7*16 for WMMA M tiles
#define EMB_D1 25
#define EMB_D2 50
#define EMB_D3 100
#define FIT_IN 1600           // 16 * 100
#define FIT_H  240

typedef __bf16 bf16;
typedef __attribute__((ext_vector_type(16))) __bf16 v16bf;
typedef __attribute__((ext_vector_type(8)))  __bf16 v8bf;
typedef __attribute__((ext_vector_type(2)))  __bf16 v2bf;
typedef __attribute__((ext_vector_type(8)))  float  v8f;
typedef __attribute__((ext_vector_type(4)))  float  f4;

// ---------------------------------------------------------------------------
// Branch-free fast tanh: sign(x) * (1 - e^{-2|x|}) / (1 + e^{-2|x|})
// fast_tanh(0) == 0, so zero-padded inputs yield zero outputs.
// ---------------------------------------------------------------------------
__device__ __forceinline__ float fast_tanh(float x) {
  float ax = fabsf(x);
  float t  = __expf(-2.0f * ax);
  float r  = __builtin_amdgcn_rcpf(1.0f + t);
  float y  = (1.0f - t) * r;
  return copysignf(y, x);
}

// ---------------------------------------------------------------------------
// WMMA fragment helpers: v_wmma_f32_16x16x32_bf16
//   A: 16x32 (MxK)  lanes 0-15 -> M, VGPR packs K pairs (ISA 7.12.2)
//   B: 32x16 (KxN)  swizzled fragment-major in memory: [tile][lane][16]
//   C: v8f, element v -> (M = v + 8*(lane>=16), N = lane&15)
// ---------------------------------------------------------------------------
__device__ __host__ __forceinline__ int frag_kbase(int v, int half) {
  return (v < 4) ? (2 * v + 8 * half) : (16 + 2 * (v - 4) + 8 * half);
}

__device__ __forceinline__ v16bf load_frag_a(const bf16* p, int ld) {
  int lane = threadIdx.x & 31;
  int half = lane >> 4;
  int m = lane & 15;
  v16bf a;
#pragma unroll
  for (int v = 0; v < 8; ++v) {
    int k = frag_kbase(v, half);
    a[2 * v]     = p[m * ld + k];
    a[2 * v + 1] = p[m * ld + k + 1];
  }
  return a;
}

// swizzled B tile: per-lane 16 contiguous bf16 (32B) -> 2x b128 loads
__device__ __forceinline__ v16bf load_frag_b_sw(const bf16* tile) {
  int lane = threadIdx.x & 31;
  return *(const v16bf*)(tile + lane * 16);
}

__device__ __forceinline__ v8f wmma_bf16(v16bf a, v16bf b, v8f c) {
  return __builtin_amdgcn_wmma_f32_16x16x32_bf16(false, a, false, b,
                                                 (short)0, c, false, false);
}

// ---------------------------------------------------------------------------
// Kernel 0: convert fp32 weights -> swizzled (fragment-major) bf16 tiles
//   dst[t]: t = tile*512 + lane*16 + e ; tile = nt*kt + ks
// ---------------------------------------------------------------------------
struct CvtArgs {
  const float* embW2[4];   // 25x50
  const float* embW3[4];   // 50x100
  const float* fitW1[2];   // 1600x240
  const float* fitW2[2];   // 240x240
  const float* fitW3[2];   // 240x240
  bf16* embW2p[4];         // K=32  N=64  : 4 tiles
  bf16* embW3p[4];         // K=64  N=112 : 14 tiles
  bf16* fitW1p[2];         // K=1600 N=240: 750 tiles
  bf16* fitW2p[2];         // K=256 N=240 : 120 tiles
  bf16* fitW3p[2];         // K=256 N=240 : 120 tiles
};

__device__ __forceinline__ void cvt_swizzle(const float* src, bf16* dst,
                                            int K, int N, int Ksrc, int Nsrc,
                                            int t0, int stride) {
  int kt = K / 32;
  int total = kt * (N / 16) * 512;
  for (int t = t0; t < total; t += stride) {
    int tile = t >> 9, rem = t & 511;
    int lane = rem >> 4, e = rem & 15;
    int nt = tile / kt, ks = tile % kt;
    int k = ks * 32 + frag_kbase(e >> 1, lane >> 4) + (e & 1);
    int n = nt * 16 + (lane & 15);
    float val = (k < Ksrc && n < Nsrc) ? src[k * Nsrc + n] : 0.0f;
    dst[t] = (bf16)val;
  }
}

__global__ __launch_bounds__(256) void cvt_kernel(CvtArgs a) {
  int stride = gridDim.x * blockDim.x;
  int t0 = blockIdx.x * blockDim.x + threadIdx.x;
  for (int e = 0; e < 4; ++e) {
    cvt_swizzle(a.embW2[e], a.embW2p[e], 32, 64, EMB_D1, EMB_D2, t0, stride);
    cvt_swizzle(a.embW3[e], a.embW3p[e], 64, 112, EMB_D2, EMB_D3, t0, stride);
  }
  for (int i = 0; i < 2; ++i) {
    cvt_swizzle(a.fitW1[i], a.fitW1p[i], FIT_IN, FIT_H, FIT_IN, FIT_H, t0, stride);
    cvt_swizzle(a.fitW2[i], a.fitW2p[i], 256, FIT_H, FIT_H, FIT_H, t0, stride);
    cvt_swizzle(a.fitW3[i], a.fitW3p[i], 256, FIT_H, FIT_H, FIT_H, t0, stride);
  }
}

// ---------------------------------------------------------------------------
// Kernel 1: fused embedding net + einsum + DR descriptor, one atom per block
// ---------------------------------------------------------------------------
struct EmbArgs {
  const float* Ri;                      // (B, 2*NAT, 2*MAXNB, 4)
  const float* W1[4];  const float* B1[4];
  const float* B2[4];  const float* B3[4];
  const bf16*  W2p[4]; const bf16*  W3p[4];  // swizzled bf16 weights
  bf16* DR;                             // (2*B*NAT, 1600) bf16 out
};

__global__ __launch_bounds__(256) void emb_kernel(EmbArgs args) {
  __shared__ float sRij[KROWS][4];       //  1.8 KB
  __shared__ bf16  sH1[KROWS][32];       //  7.2 KB (cols 25..31 zero)
  __shared__ bf16  sH2[KROWS][64];       // 14.3 KB (cols 50..63 zero)
  __shared__ float sG[KROWS][112];       // 50.2 KB
  __shared__ float sXYZ[4][104];         //  1.7 KB (g < 100 used)
  __shared__ float sW1[32], sB1[32], sB2[64], sB3[112];

  const int tid  = threadIdx.x;
  const int wave = tid >> 5;
  const int lane = tid & 31;
  const int half = lane >> 4;
  const int nl   = lane & 15;

  int atom = blockIdx.x;                       // b * 2048 + i*1024 + n
  int n = atom % NATOMS;
  int i = (atom / NATOMS) % NTYPES;
  int b = atom / (NATOMS * NTYPES);

  for (int t = tid; t < 4 * 104; t += 256) ((float*)sXYZ)[t] = 0.0f;

  for (int j = 0; j < NTYPES; ++j) {
    const int e = i * NTYPES + j;
    const float* Rbase = args.Ri +
        (((size_t)b * (NTYPES * NATOMS) + (size_t)i * NATOMS + n) *
             (NTYPES * MAXNB) + (size_t)j * MAXNB) * 4;

    // --- load Rij tile (float4, pad rows >= 100 with zero) + biases ---
    for (int t = tid; t < KROWS; t += 256) {
      f4 val = {0.0f, 0.0f, 0.0f, 0.0f};
      if (t < MAXNB) val = *(const f4*)(Rbase + t * 4);
      *(f4*)&sRij[t][0] = val;
    }
    if (tid < 32) {
      sW1[tid] = (tid < EMB_D1) ? args.W1[e][tid] : 0.0f;
      sB1[tid] = (tid < EMB_D1) ? args.B1[e][tid] : 0.0f;
    } else if (tid < 96) {
      int c = tid - 32;
      sB2[c] = (c < EMB_D2) ? args.B2[e][c] : 0.0f;
    } else if (tid < 208) {
      int c = tid - 96;
      sB3[c] = (c < EMB_D3) ? args.B3[e][c] : 0.0f;
    }
    __syncthreads();

    // --- layer 1 (1 -> 25), VALU, branchless (padding cols are tanh(0)=0) ---
    for (int t = tid; t < KROWS * 32; t += 256) {
      int r = t >> 5, c = t & 31;
      sH1[r][c] = (bf16)fast_tanh(sRij[r][0] * sW1[c] + sB1[c]);
    }
    __syncthreads();

    // --- layer 2 (25 -> 50) : 7 M-tiles x 4 N-tiles, K = 32 ---
    for (int tile = wave; tile < 28; tile += 8) {
      int mt = tile >> 2, nt = tile & 3;
      v16bf af = load_frag_a(&sH1[mt * 16][0], 32);
      v16bf bf = load_frag_b_sw(args.W2p[e] + nt * 512);
      v8f c = {};
      c = wmma_bf16(af, bf, c);
      const int nn = nt * 16 + nl;
      const float bias = sB2[nn];                       // hoisted, 1 LDS load
      const int rc = ((nn < EMB_D1) ? nn : (nn - EMB_D1)) & 31;  // in-bounds
#pragma unroll
      for (int v = 0; v < 8; ++v) {
        int m = mt * 16 + v + 8 * half;
        float val = fast_tanh(c[v] + bias) + (float)sH1[m][rc];
        sH2[m][nn] = (bf16)((nn < EMB_D2) ? val : 0.0f); // cndmask, no branch
      }
    }
    __syncthreads();

    // --- layer 3 (50 -> 100) : 7x7 tiles, K = 64 (2 steps) ---
    for (int tile = wave; tile < 49; tile += 8) {
      int mt = tile / 7, nt = tile % 7;
      v8f c = {};
#pragma unroll
      for (int ks = 0; ks < 2; ++ks) {
        v16bf af = load_frag_a(&sH2[mt * 16][ks * 32], 64);
        v16bf bf = load_frag_b_sw(args.W3p[e] + (nt * 2 + ks) * 512);
        c = wmma_bf16(af, bf, c);
      }
      const int g = nt * 16 + nl;                       // g < 112
      const float bias = sB3[g];
      const int rc = (g < EMB_D2) ? g : (g - EMB_D2);   // <= 61 < 64, in-bounds
#pragma unroll
      for (int v = 0; v < 8; ++v) {
        int m = mt * 16 + v + 8 * half;
        float val = fast_tanh(c[v] + bias) + (float)sH2[m][rc];
        sG[m][g] = (g < EMB_D3) ? val : 0.0f;
      }
    }
    __syncthreads();

    // --- einsum bnkd,bnkg -> (d,g), float4 along g, accumulate over k<100 ---
    for (int t = tid; t < 100; t += 256) {
      int d = t & 3, q = t >> 2;
      f4 acc = {0.0f, 0.0f, 0.0f, 0.0f};
      for (int k = 0; k < MAXNB; ++k) {
        float s = sRij[k][d];
        f4 sv = {s, s, s, s};
        acc += sv * *(const f4*)&sG[k][q * 4];
      }
      *(f4*)&sXYZ[d][q * 4] += acc;
    }
    __syncthreads();
  }

  // --- DR = (xyz^T @ xyz[:, :16]) * inv^2  (scaling folded into product) ---
  const float inv  = 1.0f / (float)(MAXNB * NTYPES);
  const float inv2 = inv * inv;
  size_t aIdx = (size_t)i * (NBATCH * NATOMS) + (size_t)b * NATOMS + n;
  bf16* DRp = args.DR + aIdx * FIT_IN;
  for (int t = tid; t < FIT_IN / 2; t += 256) {
    int g = t >> 3, p = t & 7;
    float a0 = 0.0f, a1 = 0.0f;
#pragma unroll
    for (int d = 0; d < 4; ++d) {
      float xg = sXYZ[d][g];
      a0 += xg * sXYZ[d][2 * p];
      a1 += xg * sXYZ[d][2 * p + 1];
    }
    v2bf pk;
    pk[0] = (bf16)(a0 * inv2);
    pk[1] = (bf16)(a1 * inv2);
    *(v2bf*)&DRp[g * 16 + 2 * p] = pk;
  }
}

// ---------------------------------------------------------------------------
// Kernel 2: fitting net, 16 atoms of one type per block
// ---------------------------------------------------------------------------
struct FitArgs {
  const bf16* DR;                       // (2*B*NAT, 1600)
  const bf16* W1[2];                    // swizzled 1600x240
  const bf16* W2[2];                    // swizzled 256x240 (K padded)
  const bf16* W3[2];                    // swizzled 256x240 (K padded)
  const float* B1[2]; const float* B2[2]; const float* B3[2];
  const float* W4[2]; const float* B4[2];
  const float* eshift;                  // [2]
  float* out;                           // (B, 2*NAT)
};

__global__ __launch_bounds__(256) void fit_kernel(FitArgs args) {
  __shared__ bf16  sA [16][FIT_IN];     // 51.2 KB
  __shared__ bf16  sHa[16][256];        //  8 KB
  __shared__ bf16  sHb[16][256];        //  8 KB
  __shared__ float sB1f[FIT_H], sB2f[FIT_H], sB3f[FIT_H], sW4f[FIT_H];
  __shared__ float sRed[16][16];

  const int tid  = threadIdx.x;
  const int wave = tid >> 5;
  const int lane = tid & 31;
  const int half = lane >> 4;
  const int nl   = lane & 15;

  const int i   = blockIdx.y;                 // atom type
  const int a0  = blockIdx.x * 16;            // atom offset within type group

  // load DR rows for 16 atoms (16B vectors) + all biases / W4
  const bf16* DRp = args.DR + ((size_t)i * (NBATCH * NATOMS) + a0) * FIT_IN;
  for (int t = tid; t < 16 * FIT_IN / 8; t += 256) {
    int r = t / (FIT_IN / 8), c8 = t % (FIT_IN / 8);
    *(v8bf*)&sA[r][c8 * 8] = *(const v8bf*)(DRp + (size_t)t * 8);
  }
  for (int t = tid; t < FIT_H; t += 256) {
    sB1f[t] = args.B1[i][t];
    sB2f[t] = args.B2[i][t];
    sB3f[t] = args.B3[i][t];
    sW4f[t] = args.W4[i][t];
  }
  { int r = tid >> 4, cp = FIT_H + (tid & 15);   // zero K-padding cols
    sHa[r][cp] = (bf16)0.0f; sHb[r][cp] = (bf16)0.0f; }
  __syncthreads();

  // --- layer 1: 1600 -> 240 (50 K-steps, 15 N-tiles) ---
  for (int nt = wave; nt < 15; nt += 8) {
    v8f c = {};
    for (int ks = 0; ks < FIT_IN / 32; ++ks) {
      v16bf af = load_frag_a(&sA[0][ks * 32], FIT_IN);
      v16bf bf = load_frag_b_sw(args.W1[i] + (size_t)(nt * 50 + ks) * 512);
      c = wmma_bf16(af, bf, c);
    }
    const int nn = nt * 16 + nl;
    const float bias = sB1f[nn];
#pragma unroll
    for (int v = 0; v < 8; ++v) {
      int m = v + 8 * half;
      sHa[m][nn] = (bf16)fast_tanh(c[v] + bias);
    }
  }
  __syncthreads();

  // --- layer 2: 240 -> 240 + residual (K padded to 256) ---
  for (int nt = wave; nt < 15; nt += 8) {
    v8f c = {};
#pragma unroll
    for (int ks = 0; ks < 8; ++ks) {
      v16bf af = load_frag_a(&sHa[0][ks * 32], 256);
      v16bf bf = load_frag_b_sw(args.W2[i] + (nt * 8 + ks) * 512);
      c = wmma_bf16(af, bf, c);
    }
    const int nn = nt * 16 + nl;
    const float bias = sB2f[nn];
#pragma unroll
    for (int v = 0; v < 8; ++v) {
      int m = v + 8 * half;
      sHb[m][nn] = (bf16)(fast_tanh(c[v] + bias) + (float)sHa[m][nn]);
    }
  }
  __syncthreads();

  // --- layer 3: 240 -> 240 + residual ---
  for (int nt = wave; nt < 15; nt += 8) {
    v8f c = {};
#pragma unroll
    for (int ks = 0; ks < 8; ++ks) {
      v16bf af = load_frag_a(&sHb[0][ks * 32], 256);
      v16bf bf = load_frag_b_sw(args.W3[i] + (nt * 8 + ks) * 512);
      c = wmma_bf16(af, bf, c);
    }
    const int nn = nt * 16 + nl;
    const float bias = sB3f[nn];
#pragma unroll
    for (int v = 0; v < 8; ++v) {
      int m = v + 8 * half;
      sHa[m][nn] = (bf16)(fast_tanh(c[v] + bias) + (float)sHb[m][nn]);
    }
  }
  __syncthreads();

  // --- final layer: 240 -> 1 (+ b4 + eshift), VALU reduction ---
  const float b4 = args.B4[i][0];
  const float es = args.eshift[i];
  {
    int r = tid >> 4, p = tid & 15;     // 16 partials per row, 15 cols each
    float acc = 0.0f;
#pragma unroll
    for (int c0 = p * 15; c0 < p * 15 + 15; ++c0)
      acc += (float)sHa[r][c0] * sW4f[c0];
    sRed[r][p] = acc;
  }
  __syncthreads();
  if (tid < 16) {
    float acc = 0.0f;
#pragma unroll
    for (int p = 0; p < 16; ++p) acc += sRed[tid][p];
    acc += b4 + es;
    int aIdx = a0 + tid;                 // within type group [0, B*NAT)
    int bb = aIdx / NATOMS, nn = aIdx % NATOMS;
    args.out[(size_t)bb * (NTYPES * NATOMS) + (size_t)i * NATOMS + nn] = acc;
  }
}

// ---------------------------------------------------------------------------
// Host launch
// ---------------------------------------------------------------------------
extern "C" void kernel_launch(void* const* d_in, const int* in_sizes, int n_in,
                              void* d_out, int out_size, void* d_ws, size_t ws_size,
                              hipStream_t stream) {
  (void)in_sizes; (void)n_in; (void)out_size; (void)ws_size;

  // d_in layout: [0]=Ri; [1..24] emb nets (4 x {W1,b1,W2,b2,W3,b3});
  // [25..40] fit nets (2 x {W1,b1,W2,b2,W3,b3,W4,b4}); [41]=energy_shift
  const float* Ri = (const float*)d_in[0];

  // workspace carve-up (bytes, every offset a multiple of 1024)
  char* ws = (char*)d_ws;
  const size_t OFF_EMBW2 = 0;                              // 4 * 4096
  const size_t OFF_EMBW3 = OFF_EMBW2 + 4 * 32 * 64 * 2;    // 4 * 14336
  const size_t OFF_FITW1 = OFF_EMBW3 + 4 * 64 * 112 * 2;
  const size_t OFF_FITW2 = OFF_FITW1 + (size_t)2 * FIT_IN * FIT_H * 2;
  const size_t OFF_FITW3 = OFF_FITW2 + (size_t)2 * 256 * FIT_H * 2;
  const size_t OFF_DR    = OFF_FITW3 + (size_t)2 * 256 * FIT_H * 2;

  CvtArgs cv; EmbArgs ea; FitArgs fa;
  ea.Ri = Ri;
  ea.DR = (bf16*)(ws + OFF_DR);
  fa.DR = (const bf16*)(ws + OFF_DR);

  for (int e = 0; e < 4; ++e) {
    int base = 1 + e * 6;
    ea.W1[e] = (const float*)d_in[base + 0];
    ea.B1[e] = (const float*)d_in[base + 1];
    cv.embW2[e] = (const float*)d_in[base + 2];
    ea.B2[e] = (const float*)d_in[base + 3];
    cv.embW3[e] = (const float*)d_in[base + 4];
    ea.B3[e] = (const float*)d_in[base + 5];
    cv.embW2p[e] = (bf16*)(ws + OFF_EMBW2 + (size_t)e * 32 * 64 * 2);
    cv.embW3p[e] = (bf16*)(ws + OFF_EMBW3 + (size_t)e * 64 * 112 * 2);
    ea.W2p[e] = cv.embW2p[e];
    ea.W3p[e] = cv.embW3p[e];
  }
  for (int i = 0; i < 2; ++i) {
    int base = 25 + i * 8;
    cv.fitW1[i] = (const float*)d_in[base + 0];
    fa.B1[i]    = (const float*)d_in[base + 1];
    cv.fitW2[i] = (const float*)d_in[base + 2];
    fa.B2[i]    = (const float*)d_in[base + 3];
    cv.fitW3[i] = (const float*)d_in[base + 4];
    fa.B3[i]    = (const float*)d_in[base + 5];
    fa.W4[i]    = (const float*)d_in[base + 6];
    fa.B4[i]    = (const float*)d_in[base + 7];
    cv.fitW1p[i] = (bf16*)(ws + OFF_FITW1 + (size_t)i * FIT_IN * FIT_H * 2);
    cv.fitW2p[i] = (bf16*)(ws + OFF_FITW2 + (size_t)i * 256 * FIT_H * 2);
    cv.fitW3p[i] = (bf16*)(ws + OFF_FITW3 + (size_t)i * 256 * FIT_H * 2);
    fa.W1[i] = cv.fitW1p[i];
    fa.W2[i] = cv.fitW2p[i];
    fa.W3[i] = cv.fitW3p[i];
  }
  fa.eshift = (const float*)d_in[41];
  fa.out = (float*)d_out;

  cvt_kernel<<<512, 256, 0, stream>>>(cv);
  emb_kernel<<<NBATCH * NTYPES * NATOMS, 256, 0, stream>>>(ea);
  fit_kernel<<<dim3((NBATCH * NATOMS) / 16, NTYPES), 256, 0, stream>>>(fa);
}
